// APA_Module_23854248362750
// MI455X (gfx1250) — matrix operand order repaired and checked by reading, not compiled, not run on hardware
//
#include <hip/hip_runtime.h>
#include <stdint.h>

// Problem constants (B, C, H, W = 4, 64, 64, 64)
#define B_   4
#define C_   64
#define C8_  8
#define N_   4096   // H*W
#define MCHUNK 32   // keys per flash-attention chunk (= WMMA K)
#define QKSTR 16    // padded halves per position in Q/K workspace (8 data + 8 zeros)

typedef __attribute__((ext_vector_type(16))) _Float16 v16h;
typedef __attribute__((ext_vector_type(8)))  float    v8f;

union V16H {
    v16h     v;
    uint32_t u[8];
    uint4    q4[2];
};

// ---------------------------------------------------------------------------
// Stage 1a: Q/K projections.  q[b,d,n] = sum_c Wq[d,c]*x_RGB[b,c,n] + bq[d]
// Q is pre-scaled by log2(e) so the attention softmax can run in exp2 domain.
// Output layout: (B, N, 16) f16 -- 8 data halves + 8 zero halves per position,
// so WMMA operand lanes can load unconditionally (pad lanes read the zeros).
// ---------------------------------------------------------------------------
__global__ __launch_bounds__(256)
void qk_proj(const float* __restrict__ xrgb,
             const float* __restrict__ Wq, const float* __restrict__ bq,
             const float* __restrict__ Wk, const float* __restrict__ bk,
             _Float16* __restrict__ Qws, _Float16* __restrict__ Kws) {
    __shared__ float wq[C8_ * C_];
    __shared__ float wk[C8_ * C_];
    __shared__ float bqi[C8_], bki[C8_];
    int tid = threadIdx.x;
    for (int i = tid; i < C8_ * C_; i += blockDim.x) { wq[i] = Wq[i]; wk[i] = Wk[i]; }
    if (tid < C8_) { bqi[tid] = bq[tid]; bki[tid] = bk[tid]; }
    __syncthreads();

    int t = blockIdx.x * blockDim.x + tid;       // over B*N
    int b = t / N_, n = t % N_;
    const float* xp = xrgb + (size_t)b * C_ * N_ + n;

    float qa[C8_], ka[C8_];
#pragma unroll
    for (int d = 0; d < C8_; ++d) { qa[d] = bqi[d]; ka[d] = bki[d]; }
    for (int c = 0; c < C_; ++c) {
        float xv = xp[(size_t)c * N_];
#pragma unroll
        for (int d = 0; d < C8_; ++d) {
            qa[d] = fmaf(wq[d * C_ + c], xv, qa[d]);
            ka[d] = fmaf(wk[d * C_ + c], xv, ka[d]);
        }
    }
    const float LOG2E = 1.4426950408889634f;
    union { _Float16 h[8]; uint4 q; } pq, pk;
#pragma unroll
    for (int d = 0; d < C8_; ++d) {
        pq.h[d] = (_Float16)(qa[d] * LOG2E);   // scores come out in log2 domain
        pk.h[d] = (_Float16)ka[d];
    }
    uint4 z; z.x = z.y = z.z = z.w = 0u;
    *(uint4*)(Qws + (size_t)t * QKSTR)     = pq.q;
    *(uint4*)(Qws + (size_t)t * QKSTR + 8) = z;
    *(uint4*)(Kws + (size_t)t * QKSTR)     = pk.q;
    *(uint4*)(Kws + (size_t)t * QKSTR + 8) = z;
}

// ---------------------------------------------------------------------------
// Stage 1b: V projection.  v[b,c,n] = sum_c' Wv[c,c']*x[b,c',n] + bv[c]
// Output layout: (B, C, N) f16  -> contiguous in key index m (WMMA B operand).
// ---------------------------------------------------------------------------
__global__ __launch_bounds__(256)
void v_proj(const float* __restrict__ x, const float* __restrict__ Wv,
            const float* __restrict__ bv, _Float16* __restrict__ Vws) {
    int t = blockIdx.x * blockDim.x + threadIdx.x;   // over B*C*N, n fastest
    int n = t % N_;
    int c = (t / N_) % C_;
    int b = t / (N_ * C_);
    const float* xp = x + (size_t)b * C_ * N_ + n;
    const float* w  = Wv + (size_t)c * C_;           // uniform per wave -> scalar loads
    float acc = bv[c];
    for (int cc = 0; cc < C_; ++cc)
        acc = fmaf(w[cc], xp[(size_t)cc * N_], acc);
    Vws[t] = (_Float16)acc;
}

// ---------------------------------------------------------------------------
// Stage 2: flash attention, one wave per 16-query tile.
//   S^T(32x16) = Kpad(32x32) x Qpad^T(32x16)   (2x wmma; D-layout == P A-layout)
//   per-lane online softmax in exp2 domain, 1 shfl_xor(16) per stat
//   O(16x64) += P(16x32) x V(32x64)            (4x wmma)
// Manual 2x ping-pong; all loads unconditional (zero-padded Q/K rows), so the
// loop body has no exec divergence and no zero re-materialization.
// ---------------------------------------------------------------------------
__global__ __launch_bounds__(128)
void attn(const _Float16* __restrict__ Qws, const _Float16* __restrict__ Kws,
          const _Float16* __restrict__ Vws, const float* __restrict__ x,
          const float* __restrict__ alpha, float* __restrict__ out) {
    const int lane = threadIdx.x & 31;
    const int wid  = blockIdx.x * 4 + (threadIdx.x >> 5);
    const int b    = wid >> 8;            // 256 query tiles per batch
    const int q0   = (wid & 255) << 4;

    const bool lo     = lane < 16;
    const int  l16    = lane & 15;
    const int  padOff = lo ? 0 : 8;           // pad lanes read the zero halves
    const int  rowSel = ((lane & 16) >> 1);   // 0 for lanes<16, 8 for lanes>=16

    // ---- Q tile as B matrix (Q^T: d x 16 queries, d padded 8->32)
    V16H qB;
#pragma unroll
    for (int i = 4; i < 8; ++i) qB.u[i] = 0u;
    qB.q4[0] = *(const uint4*)(Qws + ((size_t)b * N_ + q0 + l16) * QKSTR + padOff);

    // per-lane base pointers (lane offset + pad offset folded in)
    const _Float16* kbl = Kws + ((size_t)b * N_ + l16) * QKSTR + padOff;
    const _Float16* vbl = Vws + (size_t)b * C_ * N_ + (size_t)l16 * N_ + (lo ? 0 : 8);

    // Ping-pong operand buffers. Upper K halves zeroed ONCE; never rewritten.
    V16H kP[2], kQ[2], vP[4], vQ[4];
#pragma unroll
    for (int i = 4; i < 8; ++i) {
        kP[0].u[i] = 0u; kP[1].u[i] = 0u;
        kQ[0].u[i] = 0u; kQ[1].u[i] = 0u;
    }

    auto loadK = [&](int m0, V16H (&kn)[2]) {
        kn[0].q4[0] = *(const uint4*)(kbl + (size_t)m0 * QKSTR);
        kn[1].q4[0] = *(const uint4*)(kbl + (size_t)(m0 + 16) * QKSTR);
    };
    auto loadV = [&](int m0, V16H (&vn)[4]) {
#pragma unroll
        for (int g = 0; g < 4; ++g) {
            const _Float16* vp = vbl + (size_t)(g * 16) * N_ + m0;
            vn[g].q4[0] = *(const uint4*)(vp);
            vn[g].q4[1] = *(const uint4*)(vp + 16);
        }
    };

    loadK(0, kP);
    loadV(0, vP);

    v8f zc = {};
    v8f acc[4];
#pragma unroll
    for (int g = 0; g < 4; ++g) acc[g] = zc;
    float m_ = -__builtin_inff();   // running max, log2 domain
    float l_ = 0.f;                 // running denominator

    auto step = [&](int m0c, V16H (&kc)[2], V16H (&vc)[4],
                    V16H (&kn)[2], V16H (&vn)[4]) {
        // ---- S^T = K x Q^T : D-layout directly matches the P A-layout
        v8f s0 = __builtin_amdgcn_wmma_f32_16x16x32_f16(false, kc[0].v, false, qB.v,
                                                        (short)0, zc, false, false);
        v8f s1 = __builtin_amdgcn_wmma_f32_16x16x32_f16(false, kc[1].v, false, qB.v,
                                                        (short)0, zc, false, false);

        // ---- prefetch next chunk into the other buffer set
        int mnext = (m0c + MCHUNK) & (N_ - 1);
        loadK(mnext, kn);
        loadV(mnext, vn);

        // ---- per-lane chunk max (this lane's 16 of the row's 32 scores)
        float cmax = fmaxf(s0[0], s1[0]);
#pragma unroll
        for (int j = 1; j < 8; ++j)
            cmax = fmaxf(cmax, fmaxf(s0[j], s1[j]));
        cmax = fmaxf(cmax, __shfl_xor(cmax, 16, 32));   // merge the two lane-halves

        float mn = fmaxf(m_, cmax);
        float sc = __builtin_amdgcn_exp2f(m_ - mn);

        float p0[8], p1[8];
        float rs = 0.f;
#pragma unroll
        for (int j = 0; j < 8; ++j) {
            p0[j] = __builtin_amdgcn_exp2f(s0[j] - mn);
            p1[j] = __builtin_amdgcn_exp2f(s1[j] - mn);
            rs += p0[j] + p1[j];
        }
        rs = rs + __shfl_xor(rs, 16, 32);
        l_ = l_ * sc + rs;
        m_ = mn;

        // ---- pack P into A-matrix registers (pure per-lane f32->f16 packs)
        V16H pa;
#pragma unroll
        for (int i = 0; i < 4; ++i) {
            union { _Float16 h[2]; uint32_t u; } t0, t1;
            t0.h[0] = (_Float16)p0[2 * i];  t0.h[1] = (_Float16)p0[2 * i + 1];
            t1.h[0] = (_Float16)p1[2 * i];  t1.h[1] = (_Float16)p1[2 * i + 1];
            pa.u[i]     = t0.u;
            pa.u[4 + i] = t1.u;
        }

        // ---- rescale accumulators: row j of D lives in VGPR j (rows j / j+8)
#pragma unroll
        for (int j = 0; j < 8; ++j) {
            float scj = __shfl(sc, j + rowSel, 32);   // stats live in lanes 0..15
#pragma unroll
            for (int g = 0; g < 4; ++g) acc[g][j] *= scj;
        }

        // ---- P x V over 4 channel groups of 16
#pragma unroll
        for (int g = 0; g < 4; ++g)
            acc[g] = __builtin_amdgcn_wmma_f32_16x16x32_f16(false, pa.v, false, vc[g].v,
                                                            (short)0, acc[g], false, false);
    };

    for (int m0 = 0; m0 < N_; m0 += 2 * MCHUNK) {
        step(m0,          kP, vP, kQ, vQ);
        step(m0 + MCHUNK, kQ, vQ, kP, vP);
    }

    // ---- epilogue: normalize, alpha*out + x
    float al  = alpha[0];
    float inv = 1.0f / l_;
#pragma unroll
    for (int j = 0; j < 8; ++j) {
        float invj = __shfl(inv, j + rowSel, 32);
        int n = q0 + j + (lo ? 0 : 8);
#pragma unroll
        for (int g = 0; g < 4; ++g) {
            int c = g * 16 + l16;
            size_t idx = ((size_t)b * C_ + c) * N_ + n;
            out[idx] = fmaf(al, acc[g][j] * invj, x[idx]);
        }
    }
}

// ---------------------------------------------------------------------------
extern "C" void kernel_launch(void* const* d_in, const int* in_sizes, int n_in,
                              void* d_out, int out_size, void* d_ws, size_t ws_size,
                              hipStream_t stream) {
    const float* x     = (const float*)d_in[0];
    const float* xrgb  = (const float*)d_in[1];
    const float* Wq    = (const float*)d_in[2];
    const float* bq    = (const float*)d_in[3];
    const float* Wk    = (const float*)d_in[4];
    const float* bk    = (const float*)d_in[5];
    const float* Wv    = (const float*)d_in[6];
    const float* bv    = (const float*)d_in[7];
    const float* alpha = (const float*)d_in[8];
    float* out = (float*)d_out;

    char* ws = (char*)d_ws;
    const size_t qkBytes = (size_t)B_ * N_ * QKSTR * sizeof(_Float16); // 512 KB each
    _Float16* Qws = (_Float16*)ws;
    _Float16* Kws = (_Float16*)(ws + qkBytes);
    _Float16* Vws = (_Float16*)(ws + 2 * qkBytes);                     // 2 MB

    qk_proj<<<(B_ * N_) / 256, 256, 0, stream>>>(xrgb, Wq, bq, Wk, bk, Qws, Kws);
    v_proj<<<(B_ * C_ * N_) / 256, 256, 0, stream>>>(x, Wv, bv, Vws);
    attn<<<(B_ * N_ / 16) / 4, 128, 0, stream>>>(Qws, Kws, Vws, x, alpha, out);
}
